// VarianceAdaptor_6356551598475
// MI455X (gfx1250) — compile-verified, matrix operand order checked by reading, not run on hardware
//
#include <hip/hip_runtime.h>
#include <hip/hip_bf16.h>

#define DEV __device__ __forceinline__

constexpr int Bn  = 32;
constexpr int TXn = 384;
constexpr int TYn = 1536;
constexpr int FP  = 96;            // padded feature width for the 80-dim embeddings
#define NEGV (-1e9f)

typedef __attribute__((ext_vector_type(16))) __bf16 v16bf;
typedef __attribute__((ext_vector_type(8)))  __bf16 v8bf;
typedef __attribute__((ext_vector_type(8)))  float  v8f;

DEV v8f wmma_bf16f32(v16bf a, v16bf b, v8f c) {
  // D(f32 16x16) = A(bf16 16x32) * B(bf16 32x16) + C
  return __builtin_amdgcn_wmma_f32_16x16x32_bf16(false, a, false, b, (short)0, c, false, false);
}

// gfx1250 async global->LDS copy (ASYNCcnt-tracked).
DEV void async_load_b128_to_lds(unsigned lds_addr, const void* gaddr) {
  asm volatile("global_load_async_to_lds_b128 %0, %1, off"
               :: "v"(lds_addr), "v"(gaddr) : "memory");
}
DEV void wait_asynccnt0() { asm volatile("s_wait_asynccnt 0" ::: "memory"); }

// One 16B run (8 bf16) of a WMMA fragment -> frag[o..o+7].
DEV void load_seg8(v16bf& frag, int o, const __bf16* p) {
  v8bf t = *(const v8bf*)p;
#pragma unroll
  for (int i = 0; i < 8; ++i) frag[o + i] = t[i];
}
// Guarded variant for conv row boundaries (address-select keeps load in-bounds).
DEV void load_seg8g(v16bf& frag, int o, const __bf16* p, const __bf16* safe, bool valid) {
  v8bf t = *(const v8bf*)(valid ? p : safe);
#pragma unroll
  for (int i = 0; i < 8; ++i) frag[o + i] = valid ? t[i] : (__bf16)0.0f;
}

// ---- weight packing: (NOUT, CIN, TAPS) f32 -> per-lane WMMA B-fragment order, bf16 ----
// dst[(((nt*KC + kc)*32) + lane)*16 + i] = W[kk = kc*32 + klocal(i, lane>>4)][n = nt*16 + lane&15]
__global__ void pack_weight_frag_kernel(const float* __restrict__ w, __bf16* __restrict__ dst,
                                        int cin, int cpad, int taps, int nout) {
  int idx = blockIdx.x * blockDim.x + threadIdx.x;
  const int KC = (taps * cpad) >> 5;
  const int NT = nout >> 4;
  int total = NT * KC * 512;
  if (idx >= total) return;
  int i    = idx & 15;
  int lane = (idx >> 4) & 31;
  int kc   = (idx >> 9) % KC;
  int nt   = idx / (KC * 512);
  int g = lane >> 4;
  int kk = kc * 32 + (((i < 8) ? i : i + 8) + 8 * g);
  int kpos = kk / cpad;
  int c = kk % cpad;
  int n = nt * 16 + (lane & 15);
  float v = (c < cin) ? w[((size_t)n * cin + c) * taps + kpos] : 0.0f;
  dst[idx] = (__bf16)v;
}

// ---- activation convert + zero-pad: f32 (rows, cin) -> bf16 (rows, cpad) ----
__global__ void cvt_pad_bf16_kernel(const float* __restrict__ src, __bf16* __restrict__ dst,
                                    int rows, int cin, int cpad) {
  size_t idx = (size_t)blockIdx.x * blockDim.x + threadIdx.x;
  size_t total = (size_t)rows * cpad;
  if (idx >= total) return;
  int c = (int)(idx % cpad);
  size_t r = idx / cpad;
  dst[idx] = (__bf16)((c < cin) ? src[r * cin + c] : 0.0f);
}

__global__ void zero_u32_kernel(unsigned* __restrict__ p, size_t n) {
  size_t i = (size_t)blockIdx.x * blockDim.x + threadIdx.x;
  if (i < n) p[i] = 0u;
}

// ---------------- conv1d as implicit GEMM via WMMA (all-bf16 inputs, CIN==CPAD) ----------------
// in: (B, L, CIN) bf16 (zero-padded channels); Wp: packed fragments; out: (B, L, OSTR) bf16
// ACT: 0 none, 1 relu, 2 leaky-relu(0.01). MASKED: zero output rows >= lens[b].
template <int CIN, int TAPS, int PADDING, int NOUT, int OSTR, int ACT, int MASKED>
__launch_bounds__(32)
__global__ void conv_wmma_kernel(const __bf16* __restrict__ in, const __bf16* __restrict__ Wp,
                                 const float* __restrict__ bias, __bf16* __restrict__ out,
                                 int L, const int* __restrict__ lens) {
  constexpr int KC = (TAPS * CIN) >> 5;
  const int lane = threadIdx.x & 31;
  const int g = lane >> 4;
  const int r = lane & 15;
  const int mt = blockIdx.x, nt = blockIdx.y, b = blockIdx.z;
  const int x = mt * 16 + r;   // A-matrix row (output position)
  const int n = nt * 16 + r;   // B/C column (output channel)
  const size_t inb = (size_t)b * L * CIN;
  const __bf16* wbase = Wp + ((size_t)nt * KC * 32 + lane) * 16;
  v8f acc;
  {
    float bs = bias[n];
#pragma unroll
    for (int i = 0; i < 8; ++i) acc[i] = bs;
  }
#pragma unroll
  for (int kpos = 0; kpos < TAPS; ++kpos) {
    const int xs = x + kpos - PADDING;
    const bool xin = (xs >= 0) && (xs < L);
    const __bf16* arow = in + inb + (size_t)(xin ? xs : 0) * CIN;
#pragma unroll
    for (int c0 = 0; c0 < CIN; c0 += 32) {
      const int kc = (kpos * CIN + c0) >> 5;
      v16bf a, bm;
      bm = *(const v16bf*)(wbase + (size_t)kc * 512);     // 32B contiguous per lane
      const int k0 = c0 + 8 * g;
      const int k1 = c0 + 16 + 8 * g;
      if (TAPS == 1) {            // PADDING==0: rows always in range, channels fully valid
        load_seg8(a, 0, arow + k0);
        load_seg8(a, 8, arow + k1);
      } else {                    // only the row-boundary guard survives
        load_seg8g(a, 0, arow + k0, arow, xin);
        load_seg8g(a, 8, arow + k1, arow, xin);
      }
      acc = wmma_bf16f32(a, bm, acc);
    }
  }
  const size_t outb = (size_t)b * L * OSTR;
#pragma unroll
  for (int vr = 0; vr < 8; ++vr) {
    const int xo = mt * 16 + vr + 8 * g;   // C layout: VGPR vr holds M = vr + 8*(lane>>4)
    float v = acc[vr];
    if (ACT == 1) v = fmaxf(v, 0.0f);
    if (ACT == 2) v = (v > 0.0f) ? v : 0.01f * v;
    if (MASKED) { if (xo >= lens[b]) v = 0.0f; }
    out[outb + (size_t)xo * OSTR + n] = (__bf16)v;
  }
}

// ---------------- squared row norms over the padded 96-dim rows (pad == 0) ----------------
__global__ void rownorm2_kernel(const __bf16* __restrict__ v, float* __restrict__ out, int nrows) {
  int row = blockIdx.x * blockDim.x + threadIdx.x;
  if (row >= nrows) return;
  const v8bf* p = (const v8bf*)(v + (size_t)row * FP);
  float s = 0.0f;
#pragma unroll
  for (int c = 0; c < FP / 8; ++c) {
    v8bf t = p[c];
#pragma unroll
    for (int i = 0; i < 8; ++i) { float f = (float)t[i]; s += f * f; }
  }
  out[row] = s;
}

// ---------------- fused cdist + masked softmax over TX, per (b, 16-ty-column) block ----------------
__launch_bounds__(256)
__global__ void cdist_softmax_kernel(const __bf16* __restrict__ te, const __bf16* __restrict__ me,
                                     const float* __restrict__ te2, const float* __restrict__ me2,
                                     const int* __restrict__ src_len, const int* __restrict__ mel_len,
                                     float* __restrict__ softA) {
  const int b = blockIdx.y;
  const int ty0 = blockIdx.x * 16;
  const int tid = threadIdx.x;
  const int wave = tid >> 5;
  const int lane = tid & 31;
  const int g = lane >> 4;
  const int r = lane & 15;
  __shared__ float lg[TXn * 16];
  __shared__ float red[512];
  const int slen = src_len[b], mlen = mel_len[b];
  const __bf16* teb = te + (size_t)b * TXn * FP;
  const __bf16* meb = me + (size_t)b * TYn * FP;
  v8f acc[3];
#pragma unroll
  for (int t = 0; t < 3; ++t)
#pragma unroll
    for (int i = 0; i < 8; ++i) acc[t][i] = 0.0f;
  // G = te[b] (384x96) * me[b,ty0:ty0+16,:]^T via WMMA; pads are stored zeros -> no guards
#pragma unroll
  for (int c0 = 0; c0 < FP; c0 += 32) {
    const int k0 = c0 + 8 * g;
    const int k1 = c0 + 16 + 8 * g;
    const __bf16* brow = meb + (size_t)(ty0 + r) * FP;
    v16bf bm;
    load_seg8(bm, 0, brow + k0);
    load_seg8(bm, 8, brow + k1);
#pragma unroll
    for (int t = 0; t < 3; ++t) {                 // each wave owns M-tiles wave, wave+8, wave+16
      const int mt = wave + 8 * t;
      const __bf16* arow = teb + (size_t)(mt * 16 + r) * FP;
      v16bf a;
      load_seg8(a, 0, arow + k0);
      load_seg8(a, 8, arow + k1);
      acc[t] = wmma_bf16f32(a, bm, acc[t]);
    }
  }
  const int tyc = ty0 + r;
  const float m2v = me2[(size_t)b * TYn + tyc];
  const bool tyok = tyc < mlen;
#pragma unroll
  for (int t = 0; t < 3; ++t) {
    const int mt = wave + 8 * t;
#pragma unroll
    for (int vr = 0; vr < 8; ++vr) {
      const int x = mt * 16 + vr + 8 * g;
      const float d2 = te2[(size_t)b * TXn + x] + m2v - 2.0f * acc[t][vr];
      const float dist = sqrtf(fmaxf(d2, 0.0f));
      lg[x * 16 + r] = (tyok && x < slen) ? -dist : NEGV;
    }
  }
  __syncthreads();
  // masked softmax over x (384 rows) for each of the 16 columns; 16 threads per column
  const int col = tid & 15;
  const int seg = tid >> 4;
  float mx = NEGV;
#pragma unroll
  for (int jj = 0; jj < 24; ++jj) mx = fmaxf(mx, lg[(seg * 24 + jj) * 16 + col]);
  red[seg * 16 + col] = mx;
  __syncthreads();
  float Mx = NEGV;
#pragma unroll
  for (int s = 0; s < 16; ++s) Mx = fmaxf(Mx, red[s * 16 + col]);
  const int tyc2 = ty0 + col;
  const bool ty2ok = tyc2 < mlen;
  float ev[24];
  float sum = 0.0f;
#pragma unroll
  for (int jj = 0; jj < 24; ++jj) {
    const int x = seg * 24 + jj;
    float e = (ty2ok && x < slen) ? __expf(lg[x * 16 + col] - Mx) : 0.0f;
    ev[jj] = e;
    sum += e;
  }
  red[256 + seg * 16 + col] = sum;
  __syncthreads();
  float Sm = 0.0f;
#pragma unroll
  for (int s = 0; s < 16; ++s) Sm += red[256 + s * 16 + col];
  const float inv = 1.0f / fmaxf(Sm, 1e-9f);
  const size_t base = (size_t)b * TXn * TYn + tyc2;
#pragma unroll
  for (int jj = 0; jj < 24; ++jj) {
    const int x = seg * 24 + jj;
    softA[base + (size_t)x * TYn] = ev[jj] * inv;
  }
}

// ---------------- hard_A zero + perf_mask ----------------
__global__ void mask_init_kernel(const int* __restrict__ src_len, const int* __restrict__ mel_len,
                                 float* __restrict__ hardA, float* __restrict__ perf) {
  size_t id = (size_t)blockIdx.x * blockDim.x + threadIdx.x;
  const size_t S = (size_t)Bn * TXn * TYn;
  if (id >= S) return;
  int ty = (int)(id % TYn);
  int x  = (int)((id / TYn) % TXn);
  int b  = (int)(id / ((size_t)TXn * TYn));
  hardA[id] = 0.0f;
  bool valid = (x < src_len[b]) && (ty < mel_len[b]);
  perf[id] = valid ? 0.0f : 1.0f;  // perf_mask = ~mask_ST
}

// ---------------- MAS forward DP (per batch element), dirs bit-packed ----------------
__launch_bounds__(TXn)
__global__ void mas_forward_kernel(const float* __restrict__ softA,
                                   const int* __restrict__ src_len, const int* __restrict__ mel_len,
                                   unsigned* __restrict__ dirbits) {
  const int b = blockIdx.x;
  const int i = threadIdx.x;               // 0..TXn-1
  __shared__ __align__(16) float tile[TXn * 16];
  __shared__ float prev[TXn];
  __shared__ unsigned bits[12];            // 384 bits packed
  const int slen = src_len[b], mlen = mel_len[b];
  if (i < 12) bits[i] = 0u;
  const float* src = softA + ((size_t)b * TXn + i) * TYn;
  const unsigned ldsrow = (unsigned)(size_t)&tile[i * 16];
  for (int j0 = 0; j0 < TYn; j0 += 16) {
    // async global->LDS staging of this thread's 16 DP values (4 x B128)
    const float* gsrc = src + j0;
#pragma unroll
    for (int q = 0; q < 4; ++q) async_load_b128_to_lds(ldsrow + 16u * q, gsrc + 4 * q);
    wait_asynccnt0();
    __syncthreads();
    for (int jj = 0; jj < 16; ++jj) {
      const int j = j0 + jj;
      if (j == 0) {
        prev[i] = (i == 0 && slen > 0 && mlen > 0) ? tile[0] : NEGV;  // only (0,0) at j=0
        __syncthreads();
        continue;
      }
      const float pv = prev[i];
      const float sh = (i == 0) ? NEGV : prev[i - 1];
      __syncthreads();
      const float val = (i < slen && j < mlen) ? tile[i * 16 + jj] : NEGV;
      prev[i] = val + fmaxf(pv, sh);
      const unsigned pred = (sh > pv) ? 1u : 0u;
      atomicOr(&bits[i >> 5], pred << (i & 31));
      __syncthreads();
      if (i < 12) {
        dirbits[((size_t)b * TYn + j) * 12 + i] = bits[i];
        bits[i] = 0u;
      }
    }
    __syncthreads();
  }
}

// ---------------- backtrack: stage dir bits in LDS, lane 0 walks TY ----------------
__launch_bounds__(256)
__global__ void mas_backtrack_kernel(const unsigned* __restrict__ dirbits,
                                     const int* __restrict__ src_len, const int* __restrict__ mel_len,
                                     float* __restrict__ hardA, float* __restrict__ dur) {
  const int b = blockIdx.x;
  extern __shared__ unsigned ldsbits[];    // TYn*12 words = 72KB
  const unsigned* src = dirbits + (size_t)b * TYn * 12;
  for (int t = threadIdx.x; t < TYn * 12; t += blockDim.x) ldsbits[t] = src[t];
  __syncthreads();
  if (threadIdx.x != 0) return;
  const int slen = src_len[b], mlen = mel_len[b];
  float* hA = hardA + (size_t)b * TXn * TYn;
  float* db = dur + (size_t)b * TXn;
  int idx = 0, cur = -1, cnt = 0;
  for (int j = TYn - 1; j >= 0; --j) {
    if (j == mlen - 1) idx = slen - 1;
    if (j < mlen) {
      hA[(size_t)idx * TYn + j] = 1.0f;
      if (idx != cur) {
        if (cur >= 0) db[cur] = (float)cnt;
        cur = idx; cnt = 0;
      }
      ++cnt;
      if (j > 0) {
        const unsigned w = ldsbits[j * 12 + (idx >> 5)];
        idx -= (int)((w >> (idx & 31)) & 1u);
      }
    }
  }
  if (cur >= 0) db[cur] = (float)cnt;
}

// ---------------- host launcher ----------------
extern "C" void kernel_launch(void* const* d_in, const int* in_sizes, int n_in,
                              void* d_out, int out_size, void* d_ws, size_t ws_size,
                              hipStream_t stream) {
  (void)in_sizes; (void)n_in; (void)out_size; (void)ws_size;
  const float* text_pro = (const float*)d_in[0];
  const float* mel      = (const float*)d_in[1];
  const int*   src_len  = (const int*)d_in[2];
  const int*   mel_len  = (const int*)d_in[3];
  const float* tw1 = (const float*)d_in[7];
  const float* tb1 = (const float*)d_in[8];
  const float* tw2 = (const float*)d_in[9];
  const float* tb2 = (const float*)d_in[10];
  const float* mw1 = (const float*)d_in[11];
  const float* mb1 = (const float*)d_in[12];
  const float* mw2 = (const float*)d_in[13];
  const float* mb2 = (const float*)d_in[14];
  const float* mw3 = (const float*)d_in[15];
  const float* mb3 = (const float*)d_in[16];

  char* ws = (char*)d_ws;
  size_t off = 0;
  auto alloc = [&](size_t bytes) -> void* {
    void* p = ws + off;
    off = (off + bytes + 255) & ~(size_t)255;
    return p;
  };
  __bf16* W1t = (__bf16*)alloc((size_t)768 * 512 * 2);
  __bf16* W2t = (__bf16*)alloc((size_t)512 * 80 * 2);
  __bf16* M1t = (__bf16*)alloc((size_t)288 * 160 * 2);
  __bf16* M2t = (__bf16*)alloc((size_t)480 * 80 * 2);
  __bf16* M3t = (__bf16*)alloc((size_t)96 * 80 * 2);
  __bf16* text_bf = (__bf16*)alloc((size_t)Bn * TXn * 256 * 2);
  __bf16* mel_bf  = (__bf16*)alloc((size_t)Bn * TYn * FP * 2);
  __bf16* te_h    = (__bf16*)alloc((size_t)Bn * TXn * 512 * 2);
  __bf16* me_h1   = (__bf16*)alloc((size_t)Bn * TYn * 160 * 2);
  __bf16* me_h2   = (__bf16*)alloc((size_t)Bn * TYn * FP * 2);
  __bf16* teb     = (__bf16*)alloc((size_t)Bn * TXn * FP * 2);
  __bf16* meb     = (__bf16*)alloc((size_t)Bn * TYn * FP * 2);
  float* te2 = (float*)alloc((size_t)Bn * TXn * 4);
  float* me2 = (float*)alloc((size_t)Bn * TYn * 4);
  unsigned* dirbits = (unsigned*)alloc((size_t)Bn * TYn * 12 * 4);

  const size_t S = (size_t)Bn * TXn * TYn;
  float* softA = (float*)d_out;
  float* hardA = softA + S;
  float* perf  = softA + 2 * S;
  float* dur   = softA + 3 * S;

  auto packLaunch = [&](const float* w, __bf16* dst, int cin, int cpad, int taps, int nout) {
    int total = (nout / 16) * ((taps * cpad) / 32) * 512;
    pack_weight_frag_kernel<<<(total + 255) / 256, 256, 0, stream>>>(w, dst, cin, cpad, taps, nout);
  };
  packLaunch(tw1, W1t, 256, 256, 3, 512);
  packLaunch(tw2, W2t, 512, 512, 1, 80);
  packLaunch(mw1, M1t, 80, FP, 3, 160);
  packLaunch(mw2, M2t, 160, 160, 3, 80);
  packLaunch(mw3, M3t, 80, FP, 1, 80);

  // one-shot activation convert/pad (conversion amortized across all N-tile re-reads)
  {
    size_t tTot = (size_t)Bn * TXn * 256;
    cvt_pad_bf16_kernel<<<(unsigned)((tTot + 255) / 256), 256, 0, stream>>>(text_pro, text_bf,
                                                                            Bn * TXn, 256, 256);
    size_t mTot = (size_t)Bn * TYn * FP;
    cvt_pad_bf16_kernel<<<(unsigned)((mTot + 255) / 256), 256, 0, stream>>>(mel, mel_bf,
                                                                            Bn * TYn, 80, FP);
  }
  // zero-fill padded destinations (pads must be real zeros for cdist / norms)
  {
    size_t n1 = (size_t)Bn * TYn * FP / 2;   // me_h2 (u32 words)
    zero_u32_kernel<<<(unsigned)((n1 + 255) / 256), 256, 0, stream>>>((unsigned*)me_h2, n1);
    size_t n2 = (size_t)Bn * TXn * FP / 2;   // teb
    zero_u32_kernel<<<(unsigned)((n2 + 255) / 256), 256, 0, stream>>>((unsigned*)teb, n2);
    size_t n3 = (size_t)Bn * TYn * FP / 2;   // meb
    zero_u32_kernel<<<(unsigned)((n3 + 255) / 256), 256, 0, stream>>>((unsigned*)meb, n3);
  }

  // text encoder: conv(256->512,k3,pad1)+ReLU, conv(512->80,k1)+src mask
  conv_wmma_kernel<256, 3, 1, 512, 512, 1, 0>
      <<<dim3(TXn / 16, 512 / 16, Bn), 32, 0, stream>>>(text_bf, W1t, tb1, te_h, TXn, src_len);
  conv_wmma_kernel<512, 1, 0, 80, FP, 0, 1>
      <<<dim3(TXn / 16, 80 / 16, Bn), 32, 0, stream>>>(te_h, W2t, tb2, teb, TXn, src_len);
  // mel encoder: conv(80->160,k3)+LReLU, conv(160->80,k3)+LReLU, conv(80->80,k1)+mel mask
  conv_wmma_kernel<FP, 3, 1, 160, 160, 2, 0>
      <<<dim3(TYn / 16, 160 / 16, Bn), 32, 0, stream>>>(mel_bf, M1t, mb1, me_h1, TYn, mel_len);
  conv_wmma_kernel<160, 3, 1, 80, FP, 2, 0>
      <<<dim3(TYn / 16, 80 / 16, Bn), 32, 0, stream>>>(me_h1, M2t, mb2, me_h2, TYn, mel_len);
  conv_wmma_kernel<FP, 1, 0, 80, FP, 0, 1>
      <<<dim3(TYn / 16, 80 / 16, Bn), 32, 0, stream>>>(me_h2, M3t, mb3, meb, TYn, mel_len);

  rownorm2_kernel<<<(Bn * TXn + 255) / 256, 256, 0, stream>>>(teb, te2, Bn * TXn);
  rownorm2_kernel<<<(Bn * TYn + 255) / 256, 256, 0, stream>>>(meb, me2, Bn * TYn);

  cdist_softmax_kernel<<<dim3(TYn / 16, Bn), 256, 0, stream>>>(teb, meb, te2, me2,
                                                               src_len, mel_len, softA);

  mask_init_kernel<<<(unsigned)((S + 255) / 256), 256, 0, stream>>>(src_len, mel_len, hardA, perf);
  zero_u32_kernel<<<(Bn * TXn + 255) / 256, 256, 0, stream>>>((unsigned*)dur, (size_t)Bn * TXn);

  mas_forward_kernel<<<Bn, TXn, 0, stream>>>(softA, src_len, mel_len, dirbits);
  mas_backtrack_kernel<<<Bn, 256, TYn * 12 * sizeof(unsigned), stream>>>(dirbits, src_len, mel_len,
                                                                         hardA, dur);
}
